// Radiomic_mamba_encoder_37520834298321
// MI455X (gfx1250) — compile-verified, matrix operand order checked by reading, not run on hardware
//
#include <hip/hip_runtime.h>
#include <hip/hip_bf16.h>
#include <math.h>

// ---------------------------------------------------------------------------
// Radiomic mamba encoder for MI455X (gfx1250, wave32, WMMA).
//   B=128, NF=1781, L=2048, DEPTH=4, DI=8, DS=32, DC=4, DTR=1
// Three dense GEMMs via v_wmma_f32_16x16x32_f16 (f32 accumulate),
// 16x64 output per wave (A-fragment reuse across 4 WMMAs).
// Mamba stack fused: one workgroup per batch row, everything LDS-resident.
//   wave (0..7) == inner channel d (DI=8), lane (0..31) == state s (DS=32).
// ---------------------------------------------------------------------------

#define BB 128
#define NF 1781
#define LL 2048
#define DEPTH 4
#define DI 8
#define DS 32
#define DC 4
#define TCH 64   // scan chunk length (LDS budget ~44KB)

typedef __attribute__((ext_vector_type(16))) _Float16 v16h;
typedef __attribute__((ext_vector_type(8)))  float    v8f;

// K index held by VGPR-pair v for this lane (ISA 7.12.2, 16-bit A 16x32):
//   lanes 0-15 : v<4 -> K=2v,2v+1 ; v>=4 -> K=16+2(v-4)   (kb=0)
//   lanes 16-31: same +8                                   (kb=8)
__device__ __forceinline__ int wmma_k_of(int v, int kb)
{
    return (v < 4) ? (kb + 2 * v) : (16 + kb + 2 * (v - 4));
}

// ---------------------------------------------------------------------------
// WMMA GEMM:  C[M x N] = op(A)[M x K] * W^T + bias,  W is (N x K) row-major.
// Each wave computes a 16x64 tile (4 x 16x16, shared A fragment).
// Main K loop is guard-free; at most one guarded 32-wide tail iteration.
// ---------------------------------------------------------------------------
template<int RELU>
__global__ __launch_bounds__(256)
void wmma_gemm_kernel(const float* __restrict__ A, const float* __restrict__ W,
                      const float* __restrict__ bias, float* __restrict__ C,
                      int M, int N, int K)
{
    constexpr int NT = 4;                       // 16-col tiles per wave
    const int wave = threadIdx.x >> 5;
    const int lane = threadIdx.x & 31;
    const int nGroupsN = N >> 6;                // 64-col groups
    const int group = blockIdx.x * 8 + wave;
    if (group >= (M >> 4) * nGroupsN) return;   // wave-uniform: EXEC all-1 for WMMA
    const int tm  = (group / nGroupsN) << 4;
    const int tn0 = (group % nGroupsN) << 6;

    const int hl = lane & 15;
    const int kb = (lane < 16) ? 0 : 8;

    const float* Arow = A + (size_t)(tm + hl) * K;
    const float* Wrow[NT];
#pragma unroll
    for (int t = 0; t < NT; ++t)
        Wrow[t] = W + (size_t)(tn0 + t * 16 + hl) * K;

    v8f c[NT];
#pragma unroll
    for (int t = 0; t < NT; ++t) c[t] = (v8f){};

    const int kmain = K & ~31;

    for (int k0 = 0; k0 < kmain; k0 += 32) {
        __builtin_prefetch(Arow + k0 + 256, 0, 3);      // global_prefetch_b8
        v16h af;
#pragma unroll
        for (int v = 0; v < 8; ++v) {
            const int kk = k0 + wmma_k_of(v, kb);
            float a0 = Arow[kk], a1 = Arow[kk + 1];
            if (RELU) { a0 = fmaxf(a0, 0.f); a1 = fmaxf(a1, 0.f); }
            af[2 * v]     = (_Float16)a0;
            af[2 * v + 1] = (_Float16)a1;
        }
#pragma unroll
        for (int t = 0; t < NT; ++t) {
            v16h bf;
#pragma unroll
            for (int v = 0; v < 8; ++v) {
                const int kk = k0 + wmma_k_of(v, kb);
                bf[2 * v]     = (_Float16)Wrow[t][kk];
                bf[2 * v + 1] = (_Float16)Wrow[t][kk + 1];
            }
            c[t] = __builtin_amdgcn_wmma_f32_16x16x32_f16(
                       false, af, false, bf, (short)0, c[t], false, false);
        }
    }

    if (kmain < K) {                            // single guarded tail iteration
        v16h af;
#pragma unroll
        for (int v = 0; v < 8; ++v) {
            const int kk = kmain + wmma_k_of(v, kb);
            float a0 = (kk < K)     ? Arow[kk]     : 0.f;
            float a1 = (kk + 1 < K) ? Arow[kk + 1] : 0.f;
            if (RELU) { a0 = fmaxf(a0, 0.f); a1 = fmaxf(a1, 0.f); }
            af[2 * v]     = (_Float16)a0;
            af[2 * v + 1] = (_Float16)a1;
        }
#pragma unroll
        for (int t = 0; t < NT; ++t) {
            v16h bf;
#pragma unroll
            for (int v = 0; v < 8; ++v) {
                const int kk = kmain + wmma_k_of(v, kb);
                bf[2 * v]     = (_Float16)((kk < K)     ? Wrow[t][kk]     : 0.f);
                bf[2 * v + 1] = (_Float16)((kk + 1 < K) ? Wrow[t][kk + 1] : 0.f);
            }
            c[t] = __builtin_amdgcn_wmma_f32_16x16x32_f16(
                       false, af, false, bf, (short)0, c[t], false, false);
        }
    }

    const int rbase = tm + ((lane < 16) ? 0 : 8);   // C/D layout: VGPR r -> row r / r+8
#pragma unroll
    for (int t = 0; t < NT; ++t) {
        const int col = tn0 + t * 16 + hl;
        const float bn = bias ? bias[col] : 0.f;
#pragma unroll
        for (int r = 0; r < 8; ++r)
            C[(size_t)(rbase + r) * N + col] = c[t][r] + bn;
    }
}

// ---------------------------------------------------------------------------
// LayerNorm over row of length n (one block per row), optional trailing ReLU.
// ---------------------------------------------------------------------------
__global__ __launch_bounds__(256)
void layernorm_kernel(const float* __restrict__ in, float* __restrict__ out,
                      const float* __restrict__ g, const float* __restrict__ b,
                      int n, int do_relu)
{
    __shared__ float red[256];
    const int row = blockIdx.x, tid = threadIdx.x;
    const float* p = in + (size_t)row * n;

    float s = 0.f;
    for (int i = tid; i < n; i += 256) s += p[i];
    red[tid] = s; __syncthreads();
    for (int o = 128; o; o >>= 1) { if (tid < o) red[tid] += red[tid + o]; __syncthreads(); }
    const float mean = red[0] / (float)n;
    __syncthreads();

    float v = 0.f;
    for (int i = tid; i < n; i += 256) { float d = p[i] - mean; v += d * d; }
    red[tid] = v; __syncthreads();
    for (int o = 128; o; o >>= 1) { if (tid < o) red[tid] += red[tid + o]; __syncthreads(); }
    const float inv = rsqrtf(red[0] / (float)n + 1e-5f);
    __syncthreads();

    for (int i = tid; i < n; i += 256) {
        float o_ = (p[i] - mean) * inv * g[i] + b[i];
        if (do_relu) o_ = fmaxf(o_, 0.f);
        out[(size_t)row * n + i] = o_;
    }
}

// ---------------------------------------------------------------------------
// Fused 4-depth Mamba stack. One block per batch row. 256 threads:
//   wave = d (DI=8 waves), lane = s (DS=32 lanes). h kept in LDS (ping-pong).
// ---------------------------------------------------------------------------
__device__ __forceinline__ float silu_f(float x) { return x / (1.f + __expf(-x)); }

__global__ __launch_bounds__(256)
void mamba_stack_kernel(float* __restrict__ hbuf,
                        const float* __restrict__ inW,  const float* __restrict__ cw,
                        const float* __restrict__ cb,   const float* __restrict__ xp,
                        const float* __restrict__ dtW,  const float* __restrict__ dtb,
                        const float* __restrict__ Al,   const float* __restrict__ Dpp,
                        const float* __restrict__ ow,   const float* __restrict__ rms)
{
    __shared__ float hA[LL];            // 8KB
    __shared__ float hB[LL];            // 8KB
    __shared__ float xcL[TCH][DI];
    __shared__ float dtL[TCH][DI];
    __shared__ float zsL[TCH][DI];
    __shared__ float yacc[TCH][DI];
    __shared__ float BmL[TCH][DS];      // 8KB
    __shared__ float CmL[TCH][DS];      // 8KB
    __shared__ float s_inW[2 * DI], s_cw[DI * DC], s_cb[DI];
    __shared__ float s_xp[(1 + 2 * DS) * DI];       // 520
    __shared__ float s_dtW[DI], s_dtb[DI], s_Al[DI * DS], s_Dp[DI], s_ow[DI];
    __shared__ float s_rw;

    const int bidx = blockIdx.x;
    const int tid  = threadIdx.x;
    const int d = tid >> 5;             // wave index == inner channel
    const int s = tid & 31;             // lane index == state

    float* gh = hbuf + (size_t)bidx * LL;
    for (int t = tid; t < LL; t += 256) hA[t] = gh[t];
    __syncthreads();

    for (int dep = 0; dep < DEPTH; ++dep) {
        // --- stage per-depth params into LDS ---
        for (int i = tid; i < (1 + 2 * DS) * DI; i += 256) s_xp[i] = xp[dep * 520 + i];
        if (tid < DI * DS) s_Al[tid] = Al[dep * DI * DS + tid];
        if (tid < 2 * DI)  s_inW[tid] = inW[dep * 2 * DI + tid];
        if (tid < DI * DC) s_cw[tid] = cw[dep * DI * DC + tid];
        if (tid < DI) {
            s_cb[tid]  = cb[dep * DI + tid];
            s_dtW[tid] = dtW[dep * DI + tid];
            s_dtb[tid] = dtb[dep * DI + tid];
            s_Dp[tid]  = Dpp[dep * DI + tid];
            s_ow[tid]  = ow[dep * DI + tid];
        }
        if (tid == 0) s_rw = rms[dep];
        __syncthreads();

        float* cur = (dep & 1) ? hB : hA;
        float* nxt = (dep & 1) ? hA : hB;

        const float Aneg = -__expf(s_Al[tid]);   // tid == d*32+s exactly
        float hstate = 0.f;                       // scan state for (d,s)
        const float rw = s_rw;

        for (int c = 0; c < LL / TCH; ++c) {
            // ---- phase 1: per-t activations (threads 0..TCH-1) ----
            if (tid < TCH) {
                const int t = c * TCH + tid;
                float rn[4];                      // rn(t-3..t); zero-pad below 0
#pragma unroll
                for (int k = 0; k < 4; ++k) {
                    const int tt = t - 3 + k;
                    if (tt < 0) { rn[k] = 0.f; }
                    else {
                        const float hv = cur[tt];
                        rn[k] = hv * rsqrtf(hv * hv + 1e-5f) * rw;
                    }
                }
                float xcv[DI], dtraw = 0.f;
#pragma unroll
                for (int dd = 0; dd < DI; ++dd) {
                    float pre = s_cb[dd] + s_inW[dd] *
                        (s_cw[dd * 4 + 0] * rn[0] + s_cw[dd * 4 + 1] * rn[1] +
                         s_cw[dd * 4 + 2] * rn[2] + s_cw[dd * 4 + 3] * rn[3]);
                    const float xc = silu_f(pre);
                    xcv[dd] = xc; xcL[tid][dd] = xc;
                    dtraw += xc * s_xp[dd];                     // row 0 of xp_W
                    zsL[tid][dd] = silu_f(rn[3] * s_inW[DI + dd]);
                }
                for (int ss = 0; ss < DS; ++ss) {
                    float bm = 0.f, cm = 0.f;
#pragma unroll
                    for (int dd = 0; dd < DI; ++dd) {
                        bm += xcv[dd] * s_xp[(1 + ss) * DI + dd];
                        cm += xcv[dd] * s_xp[(1 + DS + ss) * DI + dd];
                    }
                    BmL[tid][ss] = bm; CmL[tid][ss] = cm;
                }
#pragma unroll
                for (int dd = 0; dd < DI; ++dd) {
                    const float a = dtraw * s_dtW[dd] + s_dtb[dd];
                    dtL[tid][dd] = (a > 20.f) ? a : log1pf(__expf(a));  // softplus
                }
            }
            __syncthreads();

            // ---- phase 2: sequential selective scan; DS==wave32 reduction ----
            for (int j = 0; j < TCH; ++j) {
                const float dtv = dtL[j][d];
                const float dA  = __expf(dtv * Aneg);
                hstate = dA * hstate + dtv * BmL[j][s] * xcL[j][d];
                float contrib = hstate * CmL[j][s];
                contrib += __shfl_xor(contrib, 16, 32);
                contrib += __shfl_xor(contrib, 8, 32);
                contrib += __shfl_xor(contrib, 4, 32);
                contrib += __shfl_xor(contrib, 2, 32);
                contrib += __shfl_xor(contrib, 1, 32);
                if (s == 0) {
                    float y = contrib + s_Dp[d] * xcL[j][d];
                    y *= zsL[j][d];
                    yacc[j][d] = y * s_ow[d];
                }
            }
            __syncthreads();

            // ---- phase 3: contract DI + residual ----
            if (tid < TCH) {
                const int t = c * TCH + tid;
                float sum = 0.f;
#pragma unroll
                for (int dd = 0; dd < DI; ++dd) sum += yacc[tid][dd];
                nxt[t] = cur[t] + sum;
            }
            __syncthreads();
        }
        __syncthreads();
    }

    // DEPTH=4 -> final result resides in hA
    for (int t = tid; t < LL; t += 256) gh[t] = hA[t];
}

// ---------------------------------------------------------------------------
extern "C" void kernel_launch(void* const* d_in, const int* in_sizes, int n_in,
                              void* d_out, int out_size, void* d_ws, size_t ws_size,
                              hipStream_t stream)
{
    const float* x      = (const float*)d_in[0];
    const float* p1_W   = (const float*)d_in[1];
    const float* p1_b   = (const float*)d_in[2];
    const float* ln1_g  = (const float*)d_in[3];
    const float* ln1_b  = (const float*)d_in[4];
    const float* in_W   = (const float*)d_in[5];
    const float* conv_w = (const float*)d_in[6];
    const float* conv_b = (const float*)d_in[7];
    const float* xp_W   = (const float*)d_in[8];
    const float* dt_W   = (const float*)d_in[9];
    const float* dt_b   = (const float*)d_in[10];
    const float* A_log  = (const float*)d_in[11];
    const float* Dp     = (const float*)d_in[12];
    const float* out_W  = (const float*)d_in[13];
    const float* rms_w  = (const float*)d_in[14];
    const float* p2a_W  = (const float*)d_in[15];
    const float* p2a_b  = (const float*)d_in[16];
    const float* ln2_g  = (const float*)d_in[17];
    const float* ln2_b  = (const float*)d_in[18];
    const float* p2b_W  = (const float*)d_in[19];
    const float* p2b_b  = (const float*)d_in[20];

    float* ws = (float*)d_ws;
    float* h0 = ws;                          // 128*2048 : gemm1 out
    float* h1 = h0 + (size_t)BB * LL;        // 128*2048 : LN1 out / mamba in-out
    float* g2 = h1 + (size_t)BB * LL;        // 128*512  : gemm2a out
    float* g3 = g2 + (size_t)BB * 512;       // 128*512  : LN2+relu out

    // 1) h0 = relu(x) @ p1_W^T + p1_b          (M=128, N=2048, K=1781)
    //    wave-groups: (128/16)*(2048/64) = 256 -> 32 blocks of 8 waves
    wmma_gemm_kernel<1><<<32, 256, 0, stream>>>(x, p1_W, p1_b, h0, BB, LL, NF);
    // 2) h1 = LayerNorm(h0)
    layernorm_kernel<<<BB, 256, 0, stream>>>(h0, h1, ln1_g, ln1_b, LL, 0);
    // 3) fused 4-depth Mamba stack (in-place on h1)
    mamba_stack_kernel<<<BB, 256, 0, stream>>>(h1, in_W, conv_w, conv_b, xp_W,
                                               dt_W, dt_b, A_log, Dp, out_W, rms_w);
    // 4) g2 = h1 @ p2a_W^T + p2a_b             (M=128, N=512, K=2048) : 64 groups
    wmma_gemm_kernel<0><<<8, 256, 0, stream>>>(h1, p2a_W, p2a_b, g2, BB, 512, LL);
    // 5) g3 = relu(LayerNorm(g2))
    layernorm_kernel<<<BB, 256, 0, stream>>>(g2, g3, ln2_g, ln2_b, 512, 1);
    // 6) out = g3 @ p2b_W^T + p2b_b            (M=128, N=256, K=512) : 32 groups
    wmma_gemm_kernel<0><<<4, 256, 0, stream>>>(g3, p2b_W, p2b_b,
                                               (float*)d_out, BB, 256, 512);
}
